// Block_5265629904930
// MI455X (gfx1250) — matrix operand order repaired and checked by reading, not compiled, not run on hardware
//
#include <hip/hip_runtime.h>

// ---------------------------------------------------------------------------
// MI455X (gfx1250, wave32) transformer block: LN -> attn -> +res -> LN -> MoE -> +res
// Matrix math on v_wmma_f32_16x16x32_bf16; A-tiles staged via Tensor Data Mover.
// ---------------------------------------------------------------------------

typedef __bf16 bf16;
typedef __bf16 v16bf __attribute__((ext_vector_type(16)));
typedef float  v8f   __attribute__((ext_vector_type(8)));
typedef unsigned tdm_v4u __attribute__((ext_vector_type(4)));
typedef int      tdm_v8i __attribute__((ext_vector_type(8)));
typedef int      tdm_v4i __attribute__((ext_vector_type(4)));

union Frag { v16bf v; unsigned u[8]; };

__device__ __forceinline__ float bf2f(bf16 h) {
  unsigned short s = __builtin_bit_cast(unsigned short, h);
  unsigned u = ((unsigned)s) << 16;
  return __builtin_bit_cast(float, u);
}
__device__ __forceinline__ bf16 f2bf(float f) {
  unsigned u = __builtin_bit_cast(unsigned, f);
  unsigned r = (u + 0x7FFFu + ((u >> 16) & 1u)) >> 16;
  return __builtin_bit_cast(bf16, (unsigned short)r);
}
__device__ __forceinline__ v8f wmma_bf16(const Frag& a, const Frag& b, v8f c) {
  return __builtin_amdgcn_wmma_f32_16x16x32_bf16(false, a.v, false, b.v, (short)0, c, false, false);
}

// TDM: 2D bf16 tile (tileX contiguous elements x tileY rows, row stride strideElems)
// global -> LDS at byte offset ldsOff. Descriptor per CDNA5 ISA.
__device__ __forceinline__ void tdm_load_2d_bf16(const bf16* gsrc, unsigned ldsOff,
                                                 unsigned tileX, unsigned tileY,
                                                 unsigned long long strideElems,
                                                 unsigned tensorX, unsigned tensorY) {
  unsigned long long ga = (unsigned long long)(const void*)gsrc;
  tdm_v4u g0 = { 1u,                                            // count=1 (valid D#)
                 ldsOff,                                        // lds_addr
                 (unsigned)ga,                                  // global_addr[31:0]
                 (unsigned)((ga >> 32) & 0x01FFFFFFu) | 0x80000000u };  // ga[56:32] | type=2
  tdm_v8i g1;
  g1[0] = 0x10000;                                              // data_size=1 (2 bytes)
  g1[1] = (int)((tensorX & 0xFFFFu) << 16);                     // tensor_dim0[15:0]
  g1[2] = (int)(((tensorX >> 16) & 0xFFFFu) | ((tensorY & 0xFFFFu) << 16));
  g1[3] = (int)(((tensorY >> 16) & 0xFFFFu) | ((tileX & 0xFFFFu) << 16));  // tile_dim0
  g1[4] = (int)(tileY & 0xFFFFu);                               // tile_dim1 (tile_dim2=0)
  g1[5] = (int)(unsigned)(strideElems & 0xFFFFFFFFu);           // tensor_dim0_stride lo
  g1[6] = (int)(unsigned)((strideElems >> 32) & 0xFFFFu);       // stride hi (dim1 stride=0)
  g1[7] = 0;
  tdm_v4i z4 = {0, 0, 0, 0};
#if __has_include(<hip/amd_detail/amd_gfx1250_TDM.h>)
  tdm_v8i z8 = {0, 0, 0, 0, 0, 0, 0, 0};
  __builtin_amdgcn_tensor_load_to_lds(g0, g1, z4, z4, z8, 0);   // clang-23 (6-arg)
#else
  __builtin_amdgcn_tensor_load_to_lds(g0, g1, z4, z4, 0);       // ROCm 7.2 (5-arg)
#endif
}

__device__ __forceinline__ unsigned lds_off(const void* p) {
  // generic pointer to LDS: low 32 bits are the LDS byte address (ISA §10.2)
  return (unsigned)(unsigned long long)p;
}

// problem constants
#define BB 4
#define TT 1024
#define CC 768
#define HH 12
#define DD 64
#define EE 8
#define FF 3072
#define NTOK 4096
#define CAP 1280

// ---------------------------------------------------------------------------
// fp32 -> bf16 conversion (x4 vectorized)
// ---------------------------------------------------------------------------
__launch_bounds__(256)
__global__ void k_f2bf4(const float* __restrict__ src, bf16* __restrict__ dst, long n4) {
  long i = (long)blockIdx.x * 256 + threadIdx.x;
  if (i >= n4) return;
  float4 f = ((const float4*)src)[i];
  bf16* d = dst + i * 4;
  d[0] = f2bf(f.x); d[1] = f2bf(f.y); d[2] = f2bf(f.z); d[3] = f2bf(f.w);
}

// ---------------------------------------------------------------------------
// LayerNorm: one 256-thread block per row; fp32 in, bf16 out
// ---------------------------------------------------------------------------
__launch_bounds__(256)
__global__ void k_ln(const float* __restrict__ x, const float* __restrict__ w,
                     const float* __restrict__ b, bf16* __restrict__ out) {
  __shared__ float red[256];
  const int tid = threadIdx.x;
  const float* xr = x + (long)blockIdx.x * CC;
  float s = 0.f;
  for (int c = tid; c < CC; c += 256) s += xr[c];
  red[tid] = s; __syncthreads();
  for (int off = 128; off; off >>= 1) { if (tid < off) red[tid] += red[tid + off]; __syncthreads(); }
  float mu = red[0] / CC;
  __syncthreads();
  float vs = 0.f;
  for (int c = tid; c < CC; c += 256) { float d = xr[c] - mu; vs += d * d; }
  red[tid] = vs; __syncthreads();
  for (int off = 128; off; off >>= 1) { if (tid < off) red[tid] += red[tid + off]; __syncthreads(); }
  float rstd = rsqrtf(red[0] / CC + 1e-5f);
  bf16* orow = out + (long)blockIdx.x * CC;
  for (int c = tid; c < CC; c += 256)
    orow[c] = f2bf((xr[c] - mu) * rstd * w[c] + b[c]);
}

// ---------------------------------------------------------------------------
// Generic bf16 WMMA GEMM: C[M,N] = act(A[M,K] @ B[K,N] + bias) (+residual)
// Block tile 64x128, BK=32; 8 waves: 4 (M) x 2 (N); wave tile 16x64.
// A tiles: TDM double-buffered (wave 0 issues tensor_load_to_lds for tile t+1
// while tile t is consumed; drains with s_wait_tensorcnt). B tiles: staged
// transposed in LDS so fragment reads coalesce to ds_load_b128.
// epi: 0 = bias, bf16 out | 1 = bias+GELU, bf16 out | 2 = bias+residual, f32 out
// ---------------------------------------------------------------------------
#define GBM 64
#define GBN 128
#define GBK 32
__launch_bounds__(256)
__global__ void k_gemm(const bf16* __restrict__ A, long strideA,
                       const bf16* __restrict__ B, long strideB,
                       const float* __restrict__ bias, long strideBias,
                       const float* __restrict__ resid,
                       void* __restrict__ Cout, long strideC,
                       int M, int N, int K, int epi) {
  __shared__ bf16 sA[2][GBM * GBK];   // double-buffered [m][k] row-major (TDM dest)
  __shared__ bf16 sB[GBN * GBK];      // transposed: [n][k]
  const int bz = blockIdx.z;
  A += bz * strideA; B += bz * strideB; bias += bz * strideBias;
  const int tid = threadIdx.x, lane = tid & 31, wid = tid >> 5;
  const int wm = wid & 3, wn = wid >> 2;
  const int half = lane >> 4, l16 = lane & 15;
  const int mbase = blockIdx.y * GBM, nbase = blockIdx.x * GBN;
  const int nTiles = K / GBK;
  v8f acc[4] = {};
  if (wid == 0)   // prologue: TDM load of A tile 0
    tdm_load_2d_bf16(A + (long)mbase * K, lds_off(&sA[0][0]), GBK, GBM,
                     (unsigned long long)K, (unsigned)K, (unsigned)GBM);
  for (int t = 0; t < nTiles; ++t) {
    const int k0 = t * GBK;
    if (wid == 0) {
      if (t + 1 < nTiles) {
        tdm_load_2d_bf16(A + (long)mbase * K + k0 + GBK, lds_off(&sA[(t + 1) & 1][0]),
                         GBK, GBM, (unsigned long long)K, (unsigned)K, (unsigned)GBM);
        __builtin_amdgcn_s_wait_tensorcnt(1);   // tile t landed; t+1 in flight
      } else {
        __builtin_amdgcn_s_wait_tensorcnt(0);
      }
    }
    // stage B (32x128) transposed into sB[n][k]; prefetch next tile
    #pragma unroll
    for (int it = 0; it < 2; ++it) {
      int c = tid + it * 256;
      int kk = c >> 4, nc = (c & 15) * 8;
      const bf16* src = B + (long)(k0 + kk) * N + nbase + nc;
      uint4 d = *(const uint4*)src;
      if (k0 + GBK < K) __builtin_prefetch(src + (long)GBK * N, 0, 1);
      const bf16* dv = (const bf16*)&d;
      #pragma unroll
      for (int j = 0; j < 8; ++j) sB[(nc + j) * GBK + kk] = dv[j];
    }
    __syncthreads();
    const bf16* aTile = sA[t & 1];
    Frag a;
    #pragma unroll
    for (int i = 0; i < 8; ++i) {
      int kk = 2 * i + 16 * half;                              // A-frag layout
      a.u[i] = *(const unsigned*)(aTile + (wm * 16 + l16) * GBK + kk);
    }
    #pragma unroll
    for (int j = 0; j < 4; ++j) {
      Frag bfr;
      #pragma unroll
      for (int i = 0; i < 8; ++i) {
        int kk = 2 * i + 16 * half;                              // B-frag layout
        bfr.u[i] = *(const unsigned*)(sB + (wn * 64 + j * 16 + l16) * GBK + kk);
      }
      acc[j] = wmma_bf16(a, bfr, acc[j]);
    }
    __syncthreads();
  }
  // epilogue (C layout: lanes 0-15 N=lane,M=r; lanes 16-31 N=lane-16,M=r+8)
  const int row0 = mbase + wm * 16 + 8 * half;
  #pragma unroll
  for (int j = 0; j < 4; ++j) {
    int n = nbase + wn * 64 + j * 16 + l16;
    float bv = bias[n];
    #pragma unroll
    for (int r = 0; r < 8; ++r) {
      long row = row0 + r;
      float v = acc[j][r] + bv;
      if (epi == 1) v = 0.5f * v * (1.0f + erff(v * 0.70710678118f));  // exact GELU
      if (epi == 2) {
        float* C = (float*)Cout + bz * strideC;
        C[row * N + n] = v + resid[row * N + n];
      } else {
        bf16* C = (bf16*)Cout + bz * strideC;
        C[row * N + n] = f2bf(v);
      }
    }
  }
}

// ---------------------------------------------------------------------------
// Flash attention: grid (T/128, H, B), 8 waves, 128 query rows per block.
// K/V tiles (32 keys x 64) staged in LDS (V transposed); online softmax.
// ---------------------------------------------------------------------------
__launch_bounds__(256)
__global__ void k_attn(const bf16* __restrict__ qkv, bf16* __restrict__ y) {
  __shared__ bf16 sK[32 * 64];        // [key][d]
  __shared__ bf16 sV[64 * 32];        // [d][key] (transposed)
  __shared__ bf16 sP[8][16 * 32];     // per-wave P tile [m][k]
  const int qt = blockIdx.x, h = blockIdx.y, bb = blockIdx.z;
  const int tid = threadIdx.x, lane = tid & 31, wid = tid >> 5;
  const int half = lane >> 4, l16 = lane & 15;
  const long tokBase = (long)bb * TT;
  const int qbase = qt * 128;
  // Q fragments (K=64 -> two 16x32 A-frags), pairs contiguous in global memory
  Frag qf[2];
  {
    const bf16* qptr = qkv + (tokBase + qbase + wid * 16 + l16) * (3 * CC) + h * DD;
    #pragma unroll
    for (int i = 0; i < 8; ++i) {
      int kk = 2 * i + 16 * half;
      qf[0].u[i] = *(const unsigned*)(qptr + kk);
      qf[1].u[i] = *(const unsigned*)(qptr + 32 + kk);
    }
  }
  v8f o[4] = {};
  float mrow[8], lrow[8];
  #pragma unroll
  for (int r = 0; r < 8; ++r) { mrow[r] = -1e30f; lrow[r] = 0.f; }
  const int kend = qbase + 128;
  for (int kt = 0; kt < kend; kt += 32) {
    __syncthreads();
    { // stage K and V^T
      int kr = tid >> 3, dc = (tid & 7) * 8;
      const bf16* kp = qkv + (tokBase + kt + kr) * (3 * CC) + CC + h * DD + dc;
      *(uint4*)(sK + kr * 64 + dc) = *(const uint4*)kp;
      uint4 vv = *(const uint4*)(qkv + (tokBase + kt + kr) * (3 * CC) + 2 * CC + h * DD + dc);
      const bf16* ve = (const bf16*)&vv;
      #pragma unroll
      for (int j = 0; j < 8; ++j) sV[(dc + j) * 32 + kr] = ve[j];
    }
    __syncthreads();
    // S = Q K^T  (two 16-key subtiles, two 32-wide K chunks)
    v8f s0 = {}, s1 = {};
    #pragma unroll
    for (int ks = 0; ks < 2; ++ks) {
      Frag b0, b1;
      #pragma unroll
      for (int i = 0; i < 8; ++i) {
        int kk = 2 * i + 16 * half + 32 * ks;
        b0.u[i] = *(const unsigned*)(sK + l16 * 64 + kk);
        b1.u[i] = *(const unsigned*)(sK + (16 + l16) * 64 + kk);
      }
      s0 = wmma_bf16(qf[ks], b0, s0);
      s1 = wmma_bf16(qf[ks], b1, s1);
    }
    // causal mask + scale, then online softmax per row (rows = r + 8*half)
    const int qrow0 = qbase + wid * 16 + 8 * half;
    float alpha[8];
    #pragma unroll
    for (int r = 0; r < 8; ++r) {
      int qr = qrow0 + r;
      float v0 = (kt + l16 > qr) ? -1e30f : s0[r] * 0.125f;
      float v1 = (kt + 16 + l16 > qr) ? -1e30f : s1[r] * 0.125f;
      float t = fmaxf(v0, v1);
      #pragma unroll
      for (int off = 1; off < 16; off <<= 1) t = fmaxf(t, __shfl_xor(t, off, 16));
      float mnew = fmaxf(mrow[r], t);
      float a = __expf(mrow[r] - mnew);
      float p0 = __expf(v0 - mnew), p1 = __expf(v1 - mnew);
      float ps = p0 + p1;
      #pragma unroll
      for (int off = 1; off < 16; off <<= 1) ps += __shfl_xor(ps, off, 16);
      lrow[r] = lrow[r] * a + ps;
      mrow[r] = mnew; alpha[r] = a;
      s0[r] = p0; s1[r] = p1;
    }
    // C-layout fp32 -> A-layout bf16 via per-wave LDS round trip
    bf16* P = sP[wid];
    #pragma unroll
    for (int r = 0; r < 8; ++r) {
      int m = r + 8 * half;
      P[m * 32 + l16]      = f2bf(s0[r]);
      P[m * 32 + 16 + l16] = f2bf(s1[r]);
    }
    Frag pa;
    #pragma unroll
    for (int i = 0; i < 8; ++i) {
      int kk = 2 * i + 16 * half;
      pa.u[i] = *(const unsigned*)(P + l16 * 32 + kk);
    }
    #pragma unroll
    for (int j = 0; j < 4; ++j) {
      Frag vb;
      #pragma unroll
      for (int i = 0; i < 8; ++i) {
        int kk = 2 * i + 16 * half;
        vb.u[i] = *(const unsigned*)(sV + (j * 16 + l16) * 32 + kk);
      }
      #pragma unroll
      for (int r = 0; r < 8; ++r) o[j][r] *= alpha[r];
      o[j] = wmma_bf16(pa, vb, o[j]);
    }
  }
  #pragma unroll
  for (int j = 0; j < 4; ++j)
    #pragma unroll
    for (int r = 0; r < 8; ++r) {
      long row = qbase + wid * 16 + r + 8 * half;
      y[(tokBase + row) * CC + h * DD + j * 16 + l16] = f2bf(o[j][r] / lrow[r]);
    }
}

// ---------------------------------------------------------------------------
// Router logits: logits[n][e] = ln2[n,:] . w_g[:,e]
// ---------------------------------------------------------------------------
__launch_bounds__(256)
__global__ void k_router(const bf16* __restrict__ ln2, const float* __restrict__ wg,
                         float* __restrict__ logits) {
  int t = blockIdx.x * 32 + (threadIdx.x >> 3);
  int e = threadIdx.x & 7;
  const bf16* xr = ln2 + (long)t * CC;
  float s = 0.f;
  for (int c = 0; c < CC; ++c) s += bf2f(xr[c]) * wg[c * EE + e];
  logits[t * EE + e] = s;
}

// top-2 (first-occurrence tie break, matching lax.top_k) + softmax over selected
__launch_bounds__(256)
__global__ void k_topk(const float* __restrict__ logits, int* __restrict__ e0,
                       int* __restrict__ e1, float* __restrict__ w0, float* __restrict__ w1) {
  int n = blockIdx.x * 256 + threadIdx.x;
  if (n >= NTOK) return;
  const float* L = logits + n * EE;
  int i0 = 0; float v0 = L[0];
  for (int e = 1; e < EE; ++e) if (L[e] > v0) { v0 = L[e]; i0 = e; }
  int i1 = -1; float v1 = -1e38f;
  for (int e = 0; e < EE; ++e) { if (e == i0) continue; if (L[e] > v1) { v1 = L[e]; i1 = e; } }
  float p1 = __expf(v1 - v0);       // exp(v0-v0)=1
  float z = 1.f + p1;
  e0[n] = i0; e1[n] = i1; w0[n] = 1.f / z; w1[n] = p1 / z;
}

// per-expert deterministic ranking (reference cumsum order: all k=0 by n, then k=1)
__launch_bounds__(256)
__global__ void k_rank(const int* __restrict__ e0, const int* __restrict__ e1,
                       int* __restrict__ r0, int* __restrict__ r1, int* __restrict__ slot) {
  __shared__ int sc[256];
  const int e = blockIdx.x, tid = threadIdx.x;
  for (int i = tid; i < CAP; i += 256) slot[e * CAP + i] = -1;
  __syncthreads();
  int base = 0;
  for (int k = 0; k < 2; ++k) {
    const int* ch = k ? e1 : e0;
    int* rk = k ? r1 : r0;
    for (int start = 0; start < NTOK; start += 256) {
      int n = start + tid;
      int f = (ch[n] == e) ? 1 : 0;
      sc[tid] = f; __syncthreads();
      for (int off = 1; off < 256; off <<= 1) {
        int v = (tid >= off) ? sc[tid - off] : 0;
        __syncthreads();
        sc[tid] += v;
        __syncthreads();
      }
      int rank = base + sc[tid] - 1;
      int total = sc[255];
      if (f) {
        rk[n] = rank;
        if (rank < CAP) slot[e * CAP + rank] = n;
      }
      __syncthreads();
      base += total;
    }
  }
}

// gather dispatch: xb[e,slot,:] = ln2[token,:]  (zeros for empty slots)
__launch_bounds__(256)
__global__ void k_dispatch(const int* __restrict__ slot, const bf16* __restrict__ ln2,
                           bf16* __restrict__ xb) {
  long idx = (long)blockIdx.x * 256 + threadIdx.x;   // E*CAP*(C/8) chunks
  long row = idx / (CC / 8);
  int cc = (int)(idx % (CC / 8)) * 8;
  if (row >= (long)EE * CAP) return;
  int tok = slot[row];
  uint4 d = {0u, 0u, 0u, 0u};
  if (tok >= 0) d = *(const uint4*)(ln2 + (long)tok * CC + cc);
  *(uint4*)(xb + row * CC + cc) = d;
}

// combine: out = x1 + w0*eout[e0,r0,:] + w1*eout[e1,r1,:]
__launch_bounds__(256)
__global__ void k_combine(const float* __restrict__ x1, const bf16* __restrict__ eout,
                          const int* __restrict__ e0, const int* __restrict__ e1,
                          const int* __restrict__ r0, const int* __restrict__ r1,
                          const float* __restrict__ w0, const float* __restrict__ w1,
                          float* __restrict__ out) {
  long idx = (long)blockIdx.x * 256 + threadIdx.x;   // N*(C/4)
  if (idx >= (long)NTOK * (CC / 4)) return;
  int n = (int)(idx / (CC / 4));
  int c = (int)(idx % (CC / 4)) * 4;
  float4 acc = *(const float4*)(x1 + (long)n * CC + c);
  int ra = r0[n];
  if (ra < CAP) {
    const bf16* p = eout + ((long)e0[n] * CAP + ra) * CC + c;
    float w = w0[n];
    acc.x += w * bf2f(p[0]); acc.y += w * bf2f(p[1]);
    acc.z += w * bf2f(p[2]); acc.w += w * bf2f(p[3]);
  }
  int rb = r1[n];
  if (rb < CAP) {
    const bf16* p = eout + ((long)e1[n] * CAP + rb) * CC + c;
    float w = w1[n];
    acc.x += w * bf2f(p[0]); acc.y += w * bf2f(p[1]);
    acc.z += w * bf2f(p[2]); acc.w += w * bf2f(p[3]);
  }
  *(float4*)(out + (long)n * CC + c) = acc;
}

// ---------------------------------------------------------------------------
extern "C" void kernel_launch(void* const* d_in, const int* in_sizes, int n_in,
                              void* d_out, int out_size, void* d_ws, size_t ws_size,
                              hipStream_t stream) {
  const float* x          = (const float*)d_in[0];
  const float* ln1_w      = (const float*)d_in[1];
  const float* ln1_b      = (const float*)d_in[2];
  const float* ln2_w      = (const float*)d_in[3];
  const float* ln2_b      = (const float*)d_in[4];
  const float* qkv_w      = (const float*)d_in[5];
  const float* qkv_b      = (const float*)d_in[6];
  const float* proj_w     = (const float*)d_in[7];
  const float* proj_b     = (const float*)d_in[8];
  const float* w_g        = (const float*)d_in[9];
  const float* fc_w       = (const float*)d_in[10];
  const float* fc_b       = (const float*)d_in[11];
  const float* pj_w       = (const float*)d_in[12];
  const float* pj_b       = (const float*)d_in[13];
  float* out = (float*)d_out;

  // workspace layout (256B aligned)
  char* ws = (char*)d_ws;
  size_t off = 0;
  auto alloc = [&](size_t bytes) { size_t o = off; off = (off + bytes + 255) & ~(size_t)255; return o; };
  bf16* b_ln1   = (bf16*)(ws + alloc((size_t)NTOK * CC * 2));
  bf16* b_qkvw  = (bf16*)(ws + alloc((size_t)CC * 3 * CC * 2));
  bf16* b_projw = (bf16*)(ws + alloc((size_t)CC * CC * 2));
  bf16* b_fcw   = (bf16*)(ws + alloc((size_t)EE * CC * FF * 2));
  bf16* b_pjw   = (bf16*)(ws + alloc((size_t)EE * FF * CC * 2));
  bf16* b_qkv   = (bf16*)(ws + alloc((size_t)NTOK * 3 * CC * 2));
  bf16* b_y     = (bf16*)(ws + alloc((size_t)NTOK * CC * 2));
  float* b_x1   = (float*)(ws + alloc((size_t)NTOK * CC * 4));
  bf16* b_ln2   = (bf16*)(ws + alloc((size_t)NTOK * CC * 2));
  float* b_log  = (float*)(ws + alloc((size_t)NTOK * EE * 4));
  int*  b_e0    = (int*) (ws + alloc((size_t)NTOK * 4));
  int*  b_e1    = (int*) (ws + alloc((size_t)NTOK * 4));
  int*  b_r0    = (int*) (ws + alloc((size_t)NTOK * 4));
  int*  b_r1    = (int*) (ws + alloc((size_t)NTOK * 4));
  float* b_w0   = (float*)(ws + alloc((size_t)NTOK * 4));
  float* b_w1   = (float*)(ws + alloc((size_t)NTOK * 4));
  int*  b_slot  = (int*) (ws + alloc((size_t)EE * CAP * 4));
  bf16* b_xb    = (bf16*)(ws + alloc((size_t)EE * CAP * CC * 2));
  bf16* b_h     = (bf16*)(ws + alloc((size_t)EE * CAP * FF * 2));
  bf16* b_eout  = (bf16*)(ws + alloc((size_t)EE * CAP * CC * 2));

  auto cvt = [&](const float* s, bf16* d, long n) {
    long n4 = n / 4;
    k_f2bf4<<<(int)((n4 + 255) / 256), 256, 0, stream>>>(s, d, n4);
  };
  // weight conversion fp32 -> bf16 (read once)
  cvt(qkv_w, b_qkvw, (long)CC * 3 * CC);
  cvt(proj_w, b_projw, (long)CC * CC);
  cvt(fc_w, b_fcw, (long)EE * CC * FF);
  cvt(pj_w, b_pjw, (long)EE * FF * CC);

  // LN1
  k_ln<<<NTOK, 256, 0, stream>>>(x, ln1_w, ln1_b, b_ln1);
  // QKV GEMM: [4096,768]x[768,2304]
  k_gemm<<<dim3(3 * CC / GBN, NTOK / GBM, 1), 256, 0, stream>>>(
      b_ln1, 0, b_qkvw, 0, qkv_b, 0, nullptr, b_qkv, 0, NTOK, 3 * CC, CC, 0);
  // attention
  k_attn<<<dim3(TT / 128, HH, BB), 256, 0, stream>>>(b_qkv, b_y);
  // attn proj + residual -> x1 (fp32)
  k_gemm<<<dim3(CC / GBN, NTOK / GBM, 1), 256, 0, stream>>>(
      b_y, 0, b_projw, 0, proj_b, 0, x, b_x1, 0, NTOK, CC, CC, 2);
  // LN2
  k_ln<<<NTOK, 256, 0, stream>>>(b_x1, ln2_w, ln2_b, b_ln2);
  // routing
  k_router<<<NTOK / 32, 256, 0, stream>>>(b_ln2, w_g, b_log);
  k_topk<<<NTOK / 256, 256, 0, stream>>>(b_log, b_e0, b_e1, b_w0, b_w1);
  k_rank<<<EE, 256, 0, stream>>>(b_e0, b_e1, b_r0, b_r1, b_slot);
  {
    long chunks = (long)EE * CAP * (CC / 8);
    k_dispatch<<<(int)((chunks + 255) / 256), 256, 0, stream>>>(b_slot, b_ln2, b_xb);
  }
  // expert FC (+GELU), batched over experts via grid.z
  k_gemm<<<dim3(FF / GBN, CAP / GBM, EE), 256, 0, stream>>>(
      b_xb, (long)CAP * CC, b_fcw, (long)CC * FF, fc_b, (long)FF, nullptr,
      b_h, (long)CAP * FF, CAP, FF, CC, 1);
  // expert proj
  k_gemm<<<dim3(CC / GBN, CAP / GBM, EE), 256, 0, stream>>>(
      b_h, (long)CAP * FF, b_pjw, (long)FF * CC, pj_b, (long)CC, nullptr,
      b_eout, (long)CAP * CC, CAP, CC, FF, 0);
  // combine + residual -> output
  {
    long elems = (long)NTOK * (CC / 4);
    k_combine<<<(int)((elems + 255) / 256), 256, 0, stream>>>(
        b_x1, b_eout, b_e0, b_e1, b_r0, b_r1, b_w0, b_w1, out);
  }
}
